// HybridGCN_19490561589598
// MI455X (gfx1250) — compile-verified, matrix operand order checked by reading, not run on hardware
//
#include <hip/hip_runtime.h>
#include <hip/hip_bf16.h>

// Problem sizes (fixed by the reference)
#define NN   100000
#define EE   1600000
#define FIN  128
#define DIM  64
#define NC   40
#define NHE  50000

typedef __attribute__((ext_vector_type(16))) __bf16 v16bf;
typedef __attribute__((ext_vector_type(8)))  float  v8f;

// ---------------------------------------------------------------------------
// WMMA bf16 GEMM: Y[n_rows x store_cols] = X[n_rows x K] * W[K x ncolsW]
// One wave per 16-row tile, NTILES 16-col tiles per wave, K = KCHUNKS*32.
// W is staged into LDS pre-swizzled into the exact 16-bit WMMA-B lane layout,
// so each lane reads its fragment as one contiguous 32-byte ds_load.
// ---------------------------------------------------------------------------
template<int NTILES, int KCHUNKS>
__global__ void __launch_bounds__(256)
gemm_wmma(const float* __restrict__ X, const float* __restrict__ W,
          float* __restrict__ Y, int n_rows, int ldx, int ncolsW,
          int ldy, int store_cols)
{
    __shared__ __align__(32) __bf16 Blds[KCHUNKS * NTILES * 512];

    // Stage + swizzle W (f32 -> bf16) into WMMA-B layout:
    // element e = (((kc*NTILES + nt)*32 + lane)*16 + i)
    // lane: n = (lane&15)+nt*16, grp = lane>>4
    // i:    k = kc*32 + (i<8 ? 0 : 16) + grp*8 + (i&7)
    for (int e = threadIdx.x; e < KCHUNKS * NTILES * 512; e += 256) {
        int i    = e & 15;
        int lane = (e >> 4) & 31;
        int rest = e >> 9;
        int nt   = rest % NTILES;
        int kc   = rest / NTILES;
        int grp  = lane >> 4;
        int ncol = (lane & 15) + nt * 16;
        int k    = kc * 32 + ((i < 8) ? 0 : 16) + grp * 8 + (i & 7);
        float v  = (ncol < ncolsW && k < ldx) ? W[k * ncolsW + ncol] : 0.0f;
        Blds[e] = (__bf16)v;
    }
    __syncthreads();

    const int wave    = threadIdx.x >> 5;
    const int lane    = threadIdx.x & 31;
    const int rowBase = blockIdx.x * 128 + wave * 16;
    if (rowBase >= n_rows) return;          // uniform per wave (EXEC stays all-ones)

    const int row = rowBase + (lane & 15);
    const int grp = lane >> 4;

    v8f acc[NTILES] = {};

    for (int kc = 0; kc < KCHUNKS; ++kc) {
        // A fragment: 16-bit A layout — lane holds row (lane&15),
        // i 0..7 -> K = kc*32 + grp*8 + i ; i 8..15 -> K = kc*32 + 16 + grp*8 + (i-8)
        const float* base0 = X + (size_t)row * ldx + kc * 32 + grp * 8;
        float4 f0 = ((const float4*)base0)[0];
        float4 f2 = ((const float4*)(base0 + 4))[0];
        float4 f1 = ((const float4*)(base0 + 16))[0];
        float4 f3 = ((const float4*)(base0 + 20))[0];
        v16bf a;
        a[0]  = (__bf16)f0.x; a[1]  = (__bf16)f0.y; a[2]  = (__bf16)f0.z; a[3]  = (__bf16)f0.w;
        a[4]  = (__bf16)f2.x; a[5]  = (__bf16)f2.y; a[6]  = (__bf16)f2.z; a[7]  = (__bf16)f2.w;
        a[8]  = (__bf16)f1.x; a[9]  = (__bf16)f1.y; a[10] = (__bf16)f1.z; a[11] = (__bf16)f1.w;
        a[12] = (__bf16)f3.x; a[13] = (__bf16)f3.y; a[14] = (__bf16)f3.z; a[15] = (__bf16)f3.w;

#pragma unroll
        for (int t = 0; t < NTILES; ++t) {
            v16bf b = *((const v16bf*)&Blds[((kc * NTILES + t) * 32 + lane) * 16]);
            acc[t] = __builtin_amdgcn_wmma_f32_16x16x32_bf16(
                false, a, false, b, (short)0, acc[t], false, false);
        }
    }

    // D layout: VGPR r, lane -> M = r + (lane>>4)*8, N = lane&15
    const int nloc = lane & 15;
    const int mofs = grp * 8;
#pragma unroll
    for (int t = 0; t < NTILES; ++t) {
        int col = t * 16 + nloc;
        if (col < store_cols) {
#pragma unroll
            for (int r = 0; r < 8; ++r)
                Y[(size_t)(rowBase + mofs + r) * ldy + col] = acc[t][r];
        }
    }
}

// ---------------------------------------------------------------------------
// Degree counting: D[node] += 1, B[hyperedge] += 1 over incidences
// ---------------------------------------------------------------------------
__global__ void count_deg(const int* __restrict__ hn, const int* __restrict__ he,
                          float* __restrict__ Dc, float* __restrict__ Bc)
{
    int e = blockIdx.x * blockDim.x + threadIdx.x;
    if (e >= EE) return;
    atomicAdd(&Dc[hn[e]], 1.0f);
    atomicAdd(&Bc[he[e]], 1.0f);
}

// ---------------------------------------------------------------------------
// Row scatter-add: dst[didx[e]] += src[sidx[e]] * (cnt ? 1/cnt[sidx[e]] : 1)
// 16 (or 10) threads cooperate on one incidence; float4 gathers (L2-resident)
// + 4 f32 atomics per thread.
// ---------------------------------------------------------------------------
__global__ void scatter_rows(const float* __restrict__ src, const int* __restrict__ sidx,
                             const int* __restrict__ didx, const float* __restrict__ cnt,
                             float* __restrict__ dst, int ncols, int ngroups, long long total)
{
    long long t = (long long)blockIdx.x * blockDim.x + threadIdx.x;
    if (t >= total) return;
    int e = (int)(t / ngroups);
    int g = (int)(t % ngroups);
    int s = sidx[e];
    int d = didx[e];
    float sc = 1.0f;
    if (cnt) {
        float c = cnt[s];
        sc = (c > 0.0f) ? (1.0f / c) : 0.0f;
    }
    float4 v = ((const float4*)(src + (size_t)s * ncols))[g];
    float* dp = dst + (size_t)d * ncols + g * 4;
    atomicAdd(dp + 0, v.x * sc);
    atomicAdd(dp + 1, v.y * sc);
    atomicAdd(dp + 2, v.z * sc);
    atomicAdd(dp + 3, v.w * sc);
}

// ---------------------------------------------------------------------------
// finish: data = [relu](data * Dinv[node] + bias[col]) in place
// ---------------------------------------------------------------------------
__global__ void finish_k(float* __restrict__ data, const float* __restrict__ cnt,
                         const float* __restrict__ bias, int ncols, long long total,
                         int do_relu)
{
    long long t = (long long)blockIdx.x * blockDim.x + threadIdx.x;
    if (t >= total) return;
    int node = (int)(t / ncols);
    int c    = (int)(t % ncols);
    float dv = cnt[node];
    float dinv = (dv > 0.0f) ? (1.0f / dv) : 0.0f;
    float v = data[t] * dinv + bias[c];
    if (do_relu) v = fmaxf(v, 0.0f);
    data[t] = v;
}

// ---------------------------------------------------------------------------
// prep: M2 = Wv2 @ Wo2 (40x40), c2 = bv2 @ Wo2 + bo2 (40) -> M2c[1640]
// ---------------------------------------------------------------------------
__global__ void prep_m2(const float* __restrict__ Wv2, const float* __restrict__ bv2,
                        const float* __restrict__ Wo2, const float* __restrict__ bo2,
                        float* __restrict__ M2c)
{
    int t = blockIdx.x * blockDim.x + threadIdx.x;
    if (t >= NC * NC + NC) return;
    if (t < NC * NC) {
        int i = t / NC, j = t % NC;
        float s = 0.0f;
        for (int k = 0; k < NC; ++k) s += Wv2[i * NC + k] * Wo2[k * NC + j];
        M2c[t] = s;
    } else {
        int j = t - NC * NC;
        float s = bo2[j];
        for (int k = 0; k < NC; ++k) s += bv2[k] * Wo2[k * NC + j];
        M2c[t] = s;
    }
}

// ---------------------------------------------------------------------------
// final: out = log_softmax(xh2 @ M2 + c2), one node per thread, M2/c2 in LDS
// ---------------------------------------------------------------------------
__global__ void __launch_bounds__(256)
final_ls(const float* __restrict__ xh2, const float* __restrict__ M2c,
         float* __restrict__ out, int n)
{
    __shared__ float Ms[NC * NC + NC];
    for (int t = threadIdx.x; t < NC * NC + NC; t += 256) Ms[t] = M2c[t];
    __syncthreads();

    int node = blockIdx.x * blockDim.x + threadIdx.x;
    if (node >= n) return;

    const float* xr = xh2 + (size_t)node * NC;
    __builtin_prefetch(xr, 0, 0);

    float y[NC];
#pragma unroll
    for (int j = 0; j < NC; ++j) y[j] = Ms[NC * NC + j];
    for (int k = 0; k < NC; ++k) {
        float xv = xr[k];
#pragma unroll
        for (int j = 0; j < NC; ++j) y[j] += xv * Ms[k * NC + j];
    }
    float m = y[0];
#pragma unroll
    for (int j = 1; j < NC; ++j) m = fmaxf(m, y[j]);
    float s = 0.0f;
#pragma unroll
    for (int j = 0; j < NC; ++j) s += __expf(y[j] - m);
    float lse = m + __logf(s);
    float* op = out + (size_t)node * NC;
#pragma unroll
    for (int j = 0; j < NC; ++j) op[j] = y[j] - lse;
}

// ---------------------------------------------------------------------------
extern "C" void kernel_launch(void* const* d_in, const int* in_sizes, int n_in,
                              void* d_out, int out_size, void* d_ws, size_t ws_size,
                              hipStream_t stream)
{
    const float* x   = (const float*)d_in[0];
    const int*   hn  = (const int*)d_in[2];
    const int*   he  = (const int*)d_in[3];
    const float* Wh1 = (const float*)d_in[6];
    const float* bh1 = (const float*)d_in[7];
    const float* Wh2 = (const float*)d_in[18];
    const float* bh2 = (const float*)d_in[19];
    const float* Wv2 = (const float*)d_in[24];
    const float* bv2 = (const float*)d_in[25];
    const float* Wo2 = (const float*)d_in[26];
    const float* bo2 = (const float*)d_in[27];
    float* out = (float*)d_out;
    float* ws  = (float*)d_ws;

    // workspace layout (floats)
    size_t oXW1 = 0;                         // N x 64
    size_t oME1 = oXW1 + (size_t)NN * 64;    // NHE x 64
    size_t oXH1 = oME1 + (size_t)NHE * 64;   // N x 64  (agg accum -> x_hyper in place)
    size_t oXW2 = oXH1 + (size_t)NN * 64;    // N x 40
    size_t oME2 = oXW2 + (size_t)NN * NC;    // NHE x 40
    size_t oXH2 = oME2 + (size_t)NHE * NC;   // N x 40
    size_t oD   = oXH2 + (size_t)NN * NC;    // N
    size_t oB   = oD + NN;                   // NHE
    size_t oM2  = oB + NHE;                  // 40*40 + 40

    float* XW1 = ws + oXW1;
    float* ME1 = ws + oME1;
    float* XH1 = ws + oXH1;
    float* XW2 = ws + oXW2;
    float* ME2 = ws + oME2;
    float* XH2 = ws + oXH2;
    float* Dc  = ws + oD;
    float* Bc  = ws + oB;
    float* M2c = ws + oM2;

    // zero accumulators (graph-capturable stream memsets)
    hipMemsetAsync(ME1, 0, (size_t)NHE * 64 * sizeof(float), stream);
    hipMemsetAsync(XH1, 0, (size_t)NN * 64 * sizeof(float), stream);
    hipMemsetAsync(ME2, 0, (size_t)NHE * NC * sizeof(float), stream);
    hipMemsetAsync(XH2, 0, (size_t)NN * NC * sizeof(float), stream);
    hipMemsetAsync(Dc,  0, (size_t)NN * sizeof(float), stream);
    hipMemsetAsync(Bc,  0, (size_t)NHE * sizeof(float), stream);

    // degrees
    count_deg<<<(EE + 255) / 256, 256, 0, stream>>>(hn, he, Dc, Bc);

    // layer 1: XW1 = x @ Wh1  (128x64, 4 k-chunks, 4 n-tiles)
    gemm_wmma<4, 4><<<(NN + 127) / 128, 256, 0, stream>>>(x, Wh1, XW1, NN, FIN, DIM, DIM, DIM);

    // node -> hyperedge, then hyperedge -> node (scaled by 1/B)
    long long tot64 = (long long)EE * 16;
    scatter_rows<<<(unsigned)((tot64 + 255) / 256), 256, 0, stream>>>(XW1, hn, he, nullptr, ME1, 64, 16, tot64);
    scatter_rows<<<(unsigned)((tot64 + 255) / 256), 256, 0, stream>>>(ME1, he, hn, Bc, XH1, 64, 16, tot64);

    // x_hyper = relu(acc * Dinv + bh1)
    long long n64 = (long long)NN * 64;
    finish_k<<<(unsigned)((n64 + 255) / 256), 256, 0, stream>>>(XH1, Dc, bh1, 64, n64, 1);

    // layer 2: XW2 = x_hyper @ Wh2  (64x40, 2 k-chunks, 3 n-tiles padded)
    gemm_wmma<3, 2><<<(NN + 127) / 128, 256, 0, stream>>>(XH1, Wh2, XW2, NN, DIM, NC, NC, NC);

    long long tot40 = (long long)EE * 10;
    scatter_rows<<<(unsigned)((tot40 + 255) / 256), 256, 0, stream>>>(XW2, hn, he, nullptr, ME2, NC, 10, tot40);
    scatter_rows<<<(unsigned)((tot40 + 255) / 256), 256, 0, stream>>>(ME2, he, hn, Bc, XH2, NC, 10, tot40);

    long long n40 = (long long)NN * NC;
    finish_k<<<(unsigned)((n40 + 255) / 256), 256, 0, stream>>>(XH2, Dc, bh2, NC, n40, 0);

    // fused output head: out = log_softmax(XH2 @ (Wv2@Wo2) + (bv2@Wo2+bo2))
    prep_m2<<<(NC * NC + NC + 255) / 256, 256, 0, stream>>>(Wv2, bv2, Wo2, bo2, M2c);
    final_ls<<<(NN + 255) / 256, 256, 0, stream>>>(XH2, M2c, out, NN);
}
